// Seq2Seq_52012053955014
// MI455X (gfx1250) — matrix operand order repaired and checked by reading, not compiled, not run on hardware
//
#include <hip/hip_runtime.h>
#include <hip/hip_bf16.h>
#include <stdint.h>

// Problem constants (match reference)
#define V_  32000
#define E_  256
#define H_  512
#define S_  50
#define T_  40
#define B_  32
#define G4H (4 * H_)   // 2048

typedef __attribute__((ext_vector_type(16))) __bf16 v16bf;
typedef __attribute__((ext_vector_type(8)))  float  v8f;

union Frag { v16bf v; uint4 q[2]; };

__device__ __forceinline__ unsigned short f32_to_bf16_rne(float f) {
  unsigned u = __float_as_uint(f);
  unsigned r = 0x7FFFu + ((u >> 16) & 1u);
  return (unsigned short)((u + r) >> 16);
}

__device__ __forceinline__ v8f wmma_bf16(const Frag& a, const Frag& b, v8f c) {
  return __builtin_amdgcn_wmma_f32_16x16x32_bf16(false, a.v, false, b.v,
                                                 (short)0, c, false, false);
}

// Load one 32-wide K-chunk of fragments: two A tiles (rows m, m+16) from LDS,
// one B tile from global (W row-major (N,K) => W[n][kc+8kh..+7], W[n][kc+16+8kh..+7] per lane).
__device__ __forceinline__ void ld_frags(const unsigned short* lds, int pitch,
                                         const unsigned short* __restrict__ Wrow,
                                         int m, int kh, int kc,
                                         Frag& aLo, Frag& aHi, Frag& b) {
  aLo.q[0] = *(const uint4*)&lds[m * pitch + kc + kh * 8];
  aLo.q[1] = *(const uint4*)&lds[m * pitch + kc + 16 + kh * 8];
  aHi.q[0] = *(const uint4*)&lds[(m + 16) * pitch + kc + kh * 8];
  aHi.q[1] = *(const uint4*)&lds[(m + 16) * pitch + kc + 16 + kh * 8];
  b.q[0]   = *(const uint4*)&Wrow[kc + kh * 8];
  b.q[1]   = *(const uint4*)&Wrow[kc + 16 + kh * 8];
}

// ---------------- elementwise kernels ----------------

__global__ void k_f32_to_bf16(const float* __restrict__ src,
                              unsigned short* __restrict__ dst, int n) {
  int i = blockIdx.x * blockDim.x + threadIdx.x;
  int stride = gridDim.x * blockDim.x;
  for (; i < n; i += stride) dst[i] = f32_to_bf16_rne(src[i]);
}

__global__ void k_zero_u32(unsigned int* __restrict__ p, int n) {
  int i = blockIdx.x * blockDim.x + threadIdx.x;
  int stride = gridDim.x * blockDim.x;
  for (; i < n; i += stride) p[i] = 0u;
}

// one block per token row, blockDim.x == E_
__global__ void k_embed(const int* __restrict__ tok, const float* __restrict__ emb,
                        unsigned short* __restrict__ out) {
  int row = blockIdx.x;
  int e   = threadIdx.x;
  int t   = tok[row];
  out[(size_t)row * E_ + e] = f32_to_bf16_rne(emb[(size_t)t * E_ + e]);
}

// ---------------- generic WMMA GEMM (batched, time-parallel parts) ----------------
// Cout[m][n] = sum_k A[m][k]*W[n][k] + bias[n]
// A: (M,K) bf16 row-major, W: (N,K) bf16 row-major, Cout f32 row stride ldc.
// grid = (N/128, M/32); block = 256 (8 waves); each wave: 2 M-tiles x 1 N-tile.
// K/32 must be even (K in {256, 512}).
__launch_bounds__(256)
__global__ void k_gemm_bf16(const unsigned short* __restrict__ A,
                            const unsigned short* __restrict__ W,
                            const float* __restrict__ bias,
                            float* __restrict__ Cout,
                            int ldc, int K) {
  __shared__ unsigned short ldsA[32 * (H_ + 8)];     // max K = 512
  const int tid   = threadIdx.x;
  const int lane  = tid & 31;
  const int wave  = tid >> 5;
  const int mbase = blockIdx.y << 5;
  const int n     = (blockIdx.x << 7) + (wave << 4) + (lane & 15);
  const int pitch = K + 8;

  // cooperative stage of 32 x K A tile
  {
    const int rowDw = K >> 1;
    const unsigned int* Ad = (const unsigned int*)A;
    unsigned int* Ld = (unsigned int*)ldsA;
    const int pd = pitch >> 1;
    for (int i = tid; i < 32 * rowDw; i += 256) {
      int r = i / rowDw;
      int c = i - r * rowDw;
      Ld[r * pd + c] = Ad[(size_t)(mbase + r) * rowDw + c];
    }
  }
  __syncthreads();

  const int m  = lane & 15;
  const int kh = lane >> 4;
  const unsigned short* Wrow = W + (size_t)n * K;

  v8f acc0 = {0.f,0.f,0.f,0.f,0.f,0.f,0.f,0.f};
  v8f acc1 = {0.f,0.f,0.f,0.f,0.f,0.f,0.f,0.f};

  // Rotated 2-deep software pipeline with *named* buffers (no runtime-indexed
  // arrays -> everything stays in VGPRs). Loads for the next chunk are issued
  // before the WMMAs of the current chunk.
  Frag a0L, a0H, b0, a1L, a1H, b1;
  ld_frags(ldsA, pitch, Wrow, m, kh, 0, a0L, a0H, b0);
  int kc = 32;
  for (; kc + 32 < K; kc += 64) {
    ld_frags(ldsA, pitch, Wrow, m, kh, kc, a1L, a1H, b1);
    acc0 = wmma_bf16(a0L, b0, acc0);
    acc1 = wmma_bf16(a0H, b0, acc1);
    ld_frags(ldsA, pitch, Wrow, m, kh, kc + 32, a0L, a0H, b0);
    acc0 = wmma_bf16(a1L, b1, acc0);
    acc1 = wmma_bf16(a1H, b1, acc1);
  }
  ld_frags(ldsA, pitch, Wrow, m, kh, K - 32, a1L, a1H, b1);
  acc0 = wmma_bf16(a0L, b0, acc0);
  acc1 = wmma_bf16(a0H, b0, acc1);
  acc0 = wmma_bf16(a1L, b1, acc0);
  acc1 = wmma_bf16(a1H, b1, acc1);

  const float bn = bias ? bias[n] : 0.f;
  const int mm0 = mbase + (kh << 3);
#pragma unroll
  for (int r = 0; r < 8; ++r) {
    Cout[(size_t)(mm0 + r) * ldc + n]      = acc0[r] + bn;
    Cout[(size_t)(mm0 + 16 + r) * ldc + n] = acc1[r] + bn;
  }
}

// ---------------- fused recurrent step: g = Gx[t] + h@Whh^T, then LSTM cell ----
// grid = H_/32 = 16 blocks x 256 threads. Block jb owns hidden columns
// j0..j0+31 and the four matching gate slices of N (i/f/g/o interleave), so it
// can finish the cell locally after one LDS barrier. c updated in place,
// h emitted as bf16 into hout (next step's A / next layer's input).
__launch_bounds__(256)
__global__ void k_lstm_step(const unsigned short* __restrict__ hprev, // (B_,H_) bf16
                            const unsigned short* __restrict__ Whh,   // (4H,H_) bf16
                            const float* __restrict__ Gx_t,           // (B_,4H) f32
                            float* __restrict__ c,                    // (B_,H_) f32
                            unsigned short* __restrict__ hout) {      // (B_,H_) bf16
  __shared__ unsigned short ldsH[B_ * (H_ + 8)];   // 33.3 KB
  __shared__ float gLds[B_ * 132];                 // 16.9 KB (pitch 132)
  const int tid  = threadIdx.x;
  const int lane = tid & 31;
  const int wave = tid >> 5;
  const int gate = wave >> 1;                      // 0:i 1:f 2:g 3:o
  const int j0   = blockIdx.x << 5;
  const int ncl  = ((wave & 1) << 4) + (lane & 15);   // 0..31 local col
  const int n    = gate * H_ + j0 + ncl;              // gate column in 4H
  const int pitch = H_ + 8;

  // stage full hprev (32 x 512 bf16) into LDS
  {
    const unsigned int* Hd = (const unsigned int*)hprev;
    unsigned int* Ld = (unsigned int*)ldsH;
    const int pd = pitch >> 1;                     // 260
    for (int i = tid; i < B_ * (H_ >> 1); i += 256) {
      int r  = i >> 8;                             // / 256 dwords per row
      int cc = i & 255;
      Ld[r * pd + cc] = Hd[r * (H_ >> 1) + cc];
    }
  }
  __syncthreads();

  const int m  = lane & 15;
  const int kh = lane >> 4;
  const unsigned short* Wrow = Whh + (size_t)n * H_;

  v8f acc0 = {0.f,0.f,0.f,0.f,0.f,0.f,0.f,0.f};
  v8f acc1 = {0.f,0.f,0.f,0.f,0.f,0.f,0.f,0.f};

  Frag a0L, a0H, b0, a1L, a1H, b1;
  ld_frags(ldsH, pitch, Wrow, m, kh, 0, a0L, a0H, b0);
  int kc = 32;
  for (; kc + 32 < H_; kc += 64) {
    ld_frags(ldsH, pitch, Wrow, m, kh, kc, a1L, a1H, b1);
    acc0 = wmma_bf16(a0L, b0, acc0);
    acc1 = wmma_bf16(a0H, b0, acc1);
    ld_frags(ldsH, pitch, Wrow, m, kh, kc + 32, a0L, a0H, b0);
    acc0 = wmma_bf16(a1L, b1, acc0);
    acc1 = wmma_bf16(a1H, b1, acc1);
  }
  ld_frags(ldsH, pitch, Wrow, m, kh, H_ - 32, a1L, a1H, b1);
  acc0 = wmma_bf16(a0L, b0, acc0);
  acc1 = wmma_bf16(a0H, b0, acc1);
  acc0 = wmma_bf16(a1L, b1, acc0);
  acc1 = wmma_bf16(a1H, b1, acc1);

  // park gate pre-activations (+Gx) in LDS: col layout [i|f|g|o] x 32
  const int col = (gate << 5) + ncl;
  const int mm0 = kh << 3;
#pragma unroll
  for (int r = 0; r < 8; ++r) {
    const int mA = mm0 + r, mB = mm0 + 16 + r;
    gLds[mA * 132 + col] = acc0[r] + Gx_t[(size_t)mA * G4H + n];
    gLds[mB * 132 + col] = acc1[r] + Gx_t[(size_t)mB * G4H + n];
  }
  __syncthreads();

  // LSTM cell for this block's 32x32 (row, hidden-col) patch
  for (int idx = tid; idx < B_ * 32; idx += 256) {
    const int row = idx >> 5;
    const int cl  = idx & 31;
    const float* gr = gLds + row * 132;
    float gi = gr[cl];
    float gf = gr[32 + cl];
    float gc = gr[64 + cl];
    float go = gr[96 + cl];
    float si = 1.0f / (1.0f + __expf(-gi));
    float sf = 1.0f / (1.0f + __expf(-gf));
    float so = 1.0f / (1.0f + __expf(-go));
    const int off = row * H_ + j0 + cl;
    float cn = sf * c[off] + si * tanhf(gc);
    c[off] = cn;
    hout[off] = f32_to_bf16_rne(so * tanhf(cn));
  }
}

// ---------------- host orchestration ----------------

extern "C" void kernel_launch(void* const* d_in, const int* in_sizes, int n_in,
                              void* d_out, int out_size, void* d_ws, size_t ws_size,
                              hipStream_t stream) {
  const int*   src      = (const int*)d_in[0];
  const int*   trg      = (const int*)d_in[1];
  const float* enc_emb  = (const float*)d_in[2];
  const float* enc_Wih0 = (const float*)d_in[3];
  const float* enc_Whh0 = (const float*)d_in[4];
  const float* enc_b0   = (const float*)d_in[5];
  const float* enc_Wih1 = (const float*)d_in[6];
  const float* enc_Whh1 = (const float*)d_in[7];
  const float* enc_b1   = (const float*)d_in[8];
  const float* dec_emb  = (const float*)d_in[9];
  const float* dec_Wih0 = (const float*)d_in[10];
  const float* dec_Whh0 = (const float*)d_in[11];
  const float* dec_b0   = (const float*)d_in[12];
  const float* dec_Wih1 = (const float*)d_in[13];
  const float* dec_Whh1 = (const float*)d_in[14];
  const float* dec_b1   = (const float*)d_in[15];
  const float* out_W    = (const float*)d_in[16];
  const float* out_b    = (const float*)d_in[17];
  float* out = (float*)d_out;

  // ---- workspace carve ----
  char* ws = (char*)d_ws;
  size_t off = 0;
  auto carve = [&](size_t bytes) -> void* {
    void* p = ws + off;
    off = (off + bytes + 255) & ~(size_t)255;
    return p;
  };
  auto carveU16 = [&](size_t n) { return (unsigned short*)carve(n * 2); };
  auto carveF32 = [&](size_t n) { return (float*)carve(n * 4); };

  unsigned short* wEih0 = carveU16((size_t)G4H * E_);
  unsigned short* wEhh0 = carveU16((size_t)G4H * H_);
  unsigned short* wEih1 = carveU16((size_t)G4H * H_);
  unsigned short* wEhh1 = carveU16((size_t)G4H * H_);
  unsigned short* wDih0 = carveU16((size_t)G4H * E_);
  unsigned short* wDhh0 = carveU16((size_t)G4H * H_);
  unsigned short* wDih1 = carveU16((size_t)G4H * H_);
  unsigned short* wDhh1 = carveU16((size_t)G4H * H_);
  unsigned short* wOut  = carveU16((size_t)V_ * H_);
  unsigned short* xe    = carveU16((size_t)S_ * B_ * E_);
  unsigned short* xd    = carveU16((size_t)(T_ - 1) * B_ * E_);
  unsigned short* ys0e  = carveU16((size_t)S_ * B_ * H_);
  unsigned short* ys1e  = carveU16((size_t)S_ * B_ * H_);
  unsigned short* yd0   = carveU16((size_t)(T_ - 1) * B_ * H_);
  unsigned short* yd1   = carveU16((size_t)(T_ - 1) * B_ * H_);
  unsigned short* hz    = carveU16((size_t)B_ * H_);
  float* Gx = carveF32((size_t)S_ * B_ * G4H);      // reused per layer pass
  float* c0 = carveF32((size_t)B_ * H_);
  float* c1 = carveF32((size_t)B_ * H_);

  auto blocks = [](long n, int per) {
    long b = (n + per - 1) / per;
    return (unsigned)(b > 8192 ? 8192 : b);
  };
  auto cvt = [&](const float* s, unsigned short* d, int n) {
    k_f32_to_bf16<<<blocks(n, 256), 256, 0, stream>>>(s, d, n);
  };
  auto zero = [&](void* p, long n_u32) {
    k_zero_u32<<<blocks(n_u32, 256), 256, 0, stream>>>((unsigned int*)p, (int)n_u32);
  };
  auto gemm = [&](const unsigned short* A, const unsigned short* Wm, const float* bias,
                  float* Cout, int ldc, int M, int N, int K) {
    dim3 grid((unsigned)(N / 128), (unsigned)(M / 32));
    k_gemm_bf16<<<grid, 256, 0, stream>>>(A, Wm, bias, Cout, ldc, K);
  };
  auto step = [&](const unsigned short* hprev, const unsigned short* Whh,
                  const float* Gx_t, float* c, unsigned short* hout) {
    k_lstm_step<<<H_ / 32, 256, 0, stream>>>(hprev, Whh, Gx_t, c, hout);
  };

  // ---- weight + activation conversion ----
  cvt(enc_Wih0, wEih0, G4H * E_);
  cvt(enc_Whh0, wEhh0, G4H * H_);
  cvt(enc_Wih1, wEih1, G4H * H_);
  cvt(enc_Whh1, wEhh1, G4H * H_);
  cvt(dec_Wih0, wDih0, G4H * E_);
  cvt(dec_Whh0, wDhh0, G4H * H_);
  cvt(dec_Wih1, wDih1, G4H * H_);
  cvt(dec_Whh1, wDhh1, G4H * H_);
  cvt(out_W,    wOut,  V_ * H_);

  k_embed<<<S_ * B_, E_, 0, stream>>>(src, enc_emb, xe);
  k_embed<<<(T_ - 1) * B_, E_, 0, stream>>>(trg, dec_emb, xd);  // trg[:-1]

  zero(hz, (size_t)B_ * H_ / 2);       // bf16 zeros
  zero(c0, (size_t)B_ * H_);
  zero(out, (size_t)B_ * V_);          // output row t=0 is zeros

  // ---- encoder layer 0 ----
  gemm(xe, wEih0, enc_b0, Gx, G4H, S_ * B_, G4H, E_);
  for (int t = 0; t < S_; ++t) {
    const unsigned short* hprev = (t == 0) ? hz : ys0e + (size_t)(t - 1) * B_ * H_;
    step(hprev, wEhh0, Gx + (size_t)t * B_ * G4H, c0, ys0e + (size_t)t * B_ * H_);
  }

  // ---- encoder layer 1 ----
  zero(c1, (size_t)B_ * H_);
  gemm(ys0e, wEih1, enc_b1, Gx, G4H, S_ * B_, G4H, H_);
  for (int t = 0; t < S_; ++t) {
    const unsigned short* hprev = (t == 0) ? hz : ys1e + (size_t)(t - 1) * B_ * H_;
    step(hprev, wEhh1, Gx + (size_t)t * B_ * G4H, c1, ys1e + (size_t)t * B_ * H_);
  }

  // ---- decoder layer 0 (init = encoder layer-0 final h/c; c0 carries over) ----
  const int TD = T_ - 1;
  gemm(xd, wDih0, dec_b0, Gx, G4H, TD * B_, G4H, E_);
  for (int t = 0; t < TD; ++t) {
    const unsigned short* hprev = (t == 0) ? ys0e + (size_t)(S_ - 1) * B_ * H_
                                           : yd0 + (size_t)(t - 1) * B_ * H_;
    step(hprev, wDhh0, Gx + (size_t)t * B_ * G4H, c0, yd0 + (size_t)t * B_ * H_);
  }

  // ---- decoder layer 1 (init = encoder layer-1 final h/c; c1 carries over) ----
  gemm(yd0, wDih1, dec_b1, Gx, G4H, TD * B_, G4H, H_);
  for (int t = 0; t < TD; ++t) {
    const unsigned short* hprev = (t == 0) ? ys1e + (size_t)(S_ - 1) * B_ * H_
                                           : yd1 + (size_t)(t - 1) * B_ * H_;
    step(hprev, wDhh1, Gx + (size_t)t * B_ * G4H, c1, yd1 + (size_t)t * B_ * H_);
  }

  // ---- output projection: logits -> d_out rows 1..T-1 ----
  gemm(yd1, wOut, out_b, out + (size_t)B_ * V_, V_, TD * B_, V_, H_);
}
